// MambaNet_20246475833374
// MI455X (gfx1250) — matrix operand order repaired and checked by reading, not compiled, not run on hardware
//
#include <hip/hip_runtime.h>
#include <math.h>

// ---------------- Model constants (must match reference) ----------------
#define DV    768          // D
#define EDV   1536         // ED = 2*D
#define NSV   16           // state size
#define TV    1024         // sequence length
#define BV    2            // batch
#define VV    32000        // vocab
#define NLV   2            // layers
#define CK    4            // conv kernel width
#define DTR   48           // dt rank
#define FV    80           // DTR + 2*NS
#define BT    (BV*TV)      // 2048 activation rows
#define NCH   32           // scan chunks
#define LCH   (TV/NCH)     // 32 timesteps per chunk

// ---------------- WMMA types ----------------
typedef __attribute__((ext_vector_type(16))) __bf16 bf16x16;
typedef __attribute__((ext_vector_type(8)))  float  v8f;

static __device__ __forceinline__ unsigned short f2bf(float x) {
    union { float f; unsigned int u; } c; c.f = x;
    unsigned int r = c.u + 0x7fffu + ((c.u >> 16) & 1u);   // round-to-nearest-even
    return (unsigned short)(r >> 16);
}
static __device__ __forceinline__ float sigmoidf_(float x) { return 1.0f / (1.0f + __expf(-x)); }

// ==========================================================================
// Tiled bf16 WMMA GEMM: C[M,N] = A[M,K] * W[N,K]^T (+ bias[N])
//  - 128x128 block tile, BK=32, 256 threads / 8 waves (wave32)
//  - wave tile 32x64: 2 A frags x 4 B frags -> 8 v_wmma per K-step
//  - f32 -> bf16 conversion fused into LDS staging
//  - double-buffered LDS; next tile's global loads issued before compute
//  - optional non-temporal epilogue stores for streaming output
// Requires M%128==0, N%128==0, K%32==0 (true for all call sites).
// ==========================================================================
#define GBM 128
#define GBN 128
#define GBK 32
#define LDA (GBK + 2)   // ushort row stride (17 words: conflict-free, 4B-aligned)

__global__ __launch_bounds__(256)
void mamba_gemm_wmma_bf16(const float* __restrict__ A,
                          const float* __restrict__ W,
                          const float* __restrict__ bias,
                          float* __restrict__ C,
                          int M, int N, int Kdim, int nt_store) {
    __shared__ unsigned short As[2][GBM * LDA];
    __shared__ unsigned short Ws[2][GBN * LDA];

    const int tid  = threadIdx.x;
    const int lane = tid & 31;
    const int wave = tid >> 5;
    const int g    = lane >> 4;          // half-wave lane group
    const int lm   = lane & 15;
    const int bm   = blockIdx.x * GBM;
    const int bn   = blockIdx.y * GBN;
    const int wm   = (wave & 3) * 32;    // wave M offset within block tile
    const int wn   = (wave >> 2) * 64;   // wave N offset within block tile

    // staging indices: tile is 128 rows x 8 float4; thread handles i = tid + j*256
    int srow[4], scol[4];
    #pragma unroll
    for (int j = 0; j < 4; ++j) { int i = tid + j * 256; srow[j] = i >> 3; scol[j] = (i & 7) << 2; }

    v8f acc[2][4];
    #pragma unroll
    for (int mt = 0; mt < 2; ++mt)
        #pragma unroll
        for (int nt = 0; nt < 4; ++nt) acc[mt][nt] = v8f{};

    float4 ar[4], wr4[4];
    auto gload = [&](int k0) {
        #pragma unroll
        for (int j = 0; j < 4; ++j)
            ar[j] = *(const float4*)(A + (size_t)(bm + srow[j]) * Kdim + k0 + scol[j]);
        #pragma unroll
        for (int j = 0; j < 4; ++j)
            wr4[j] = *(const float4*)(W + (size_t)(bn + srow[j]) * Kdim + k0 + scol[j]);
    };
    auto stash = [&](int buf) {
        #pragma unroll
        for (int j = 0; j < 4; ++j) {
            unsigned short* d = &As[buf][srow[j] * LDA + scol[j]];
            d[0] = f2bf(ar[j].x); d[1] = f2bf(ar[j].y); d[2] = f2bf(ar[j].z); d[3] = f2bf(ar[j].w);
        }
        #pragma unroll
        for (int j = 0; j < 4; ++j) {
            unsigned short* d = &Ws[buf][srow[j] * LDA + scol[j]];
            d[0] = f2bf(wr4[j].x); d[1] = f2bf(wr4[j].y); d[2] = f2bf(wr4[j].z); d[3] = f2bf(wr4[j].w);
        }
    };

    const int nk = Kdim / GBK;
    gload(0);
    stash(0);

    for (int kt = 0; kt < nk; ++kt) {
        const int cur = kt & 1;
        __syncthreads();
        if (kt + 1 < nk) gload((kt + 1) * GBK);   // issue next tile's loads early

        // ---- A fragments (ISA 7.12.2 16-bit A layout) ----
        union { bf16x16 v; unsigned int u[8]; } afr[2];
        #pragma unroll
        for (int mt = 0; mt < 2; ++mt) {
            const int arow = wm + mt * 16 + lm;
            #pragma unroll
            for (int v = 0; v < 8; ++v) {
                int kb = ((v & 4) ? 16 : 0) + (v & 3) * 2 + g * 8;
                afr[mt].u[v] = *(const unsigned int*)&As[cur][arow * LDA + kb];
            }
        }
        // ---- 4 B fragments, 8 WMMAs ----
        #pragma unroll
        for (int nt = 0; nt < 4; ++nt) {
            union { bf16x16 v; unsigned int u[8]; } bfr;
            const int ncol = wn + nt * 16 + lm;
            #pragma unroll
            for (int v = 0; v < 8; ++v)
                bfr.u[v] = *(const unsigned int*)&Ws[cur][ncol * LDA + ((v & 4) ? 16 : 0) + (v & 3) * 2 + g * 8];
            #pragma unroll
            for (int mt = 0; mt < 2; ++mt)
                acc[mt][nt] = __builtin_amdgcn_wmma_f32_16x16x32_bf16(
                    false, afr[mt].v, false, bfr.v, (short)0, acc[mt][nt], false, false);
        }
        if (kt + 1 < nk) stash(cur ^ 1);          // write other buffer (no conflict)
    }

    // ---- epilogue: D layout -> VGPR r holds M = r + g*8 ----
    #pragma unroll
    for (int mt = 0; mt < 2; ++mt) {
        #pragma unroll
        for (int nt = 0; nt < 4; ++nt) {
            const int col = bn + wn + nt * 16 + lm;
            const float bv = bias ? bias[col] : 0.0f;
            #pragma unroll
            for (int r = 0; r < 8; ++r) {
                const int row = bm + wm + mt * 16 + g * 8 + r;
                float val = acc[mt][nt][r] + bv;
                float* p = C + (size_t)row * N + col;
                if (nt_store) __builtin_nontemporal_store(val, p);
                else          *p = val;
            }
        }
    }
}

// ==========================================================================
// Elementwise / small kernels
// ==========================================================================
__global__ __launch_bounds__(256)
void embed_kernel(const int* __restrict__ toks, const float* __restrict__ emb,
                  float* __restrict__ x, int n) {
    int i = blockIdx.x * blockDim.x + threadIdx.x;
    if (i >= n) return;
    int t = i / DV, d = i % DV;
    x[i] = emb[(size_t)toks[t] * DV + d];
}

__global__ __launch_bounds__(256)
void conv_silu_kernel(const float* __restrict__ xz, const float* __restrict__ cw,
                      const float* __restrict__ cb, float* __restrict__ xin_act) {
    int i = blockIdx.x * blockDim.x + threadIdx.x;
    if (i >= BT * EDV) return;
    int e = i % EDV, bt = i / EDV, t = bt % TV, b = bt / TV;
    float acc = cb[e];
    #pragma unroll
    for (int k = 0; k < CK; ++k) {
        int tt = t - (CK - 1) + k;
        if (tt >= 0)
            acc += xz[((size_t)(b * TV + tt)) * (2 * EDV) + e] * cw[e * CK + k];
    }
    xin_act[i] = acc * sigmoidf_(acc);
}

__global__ __launch_bounds__(256)
void xproj_kernel(const float* __restrict__ xin, const float* __restrict__ xpw,
                  float* __restrict__ dbc) {
    int i = blockIdx.x * blockDim.x + threadIdx.x;
    if (i >= BT * FV) return;
    int f = i % FV, bt = i / FV;
    const float* xr = xin + (size_t)bt * EDV;
    const float* wr = xpw + (size_t)f * EDV;
    float acc = 0.0f;
    #pragma unroll 8
    for (int e = 0; e < EDV; ++e) acc += xr[e] * wr[e];
    dbc[i] = acc;
}

__global__ __launch_bounds__(256)
void dtproj_kernel(const float* __restrict__ dbc, const float* __restrict__ dtw,
                   const float* __restrict__ dtb, float* __restrict__ delta) {
    int i = blockIdx.x * blockDim.x + threadIdx.x;
    if (i >= BT * EDV) return;
    int e = i % EDV, bt = i / EDV;
    const float* dr = dbc + (size_t)bt * FV;
    const float* wr = dtw + (size_t)e * DTR;
    float acc = dtb[e];
    #pragma unroll
    for (int r = 0; r < DTR; ++r) acc += dr[r] * wr[r];
    delta[i] = (acc > 20.0f) ? acc : log1pf(__expf(acc));
}

// ==========================================================================
// Chunk-parallel selective scan.
// h_t = a_t*h_{t-1} + b_t with a_t = exp(delta_t*A), associative -> 3 phases:
//  phase1: per (b,e,chunk): chunk product P and zero-state partial S
//  phase2: per (b,e): 32-step carry propagation, records incoming state Hin
//  phase3: per (b,e,chunk): replay from Hin, emit y (+D*xin), gate by silu(z)
// Thread index i = c*(B*ED) + be so global reads stay coalesced over e.
// ==========================================================================
__global__ __launch_bounds__(256)
void scan_phase1(const float* __restrict__ delta, const float* __restrict__ xin,
                 const float* __restrict__ dbc, const float* __restrict__ Alog,
                 float* __restrict__ P, float* __restrict__ S) {
    int i = blockIdx.x * blockDim.x + threadIdx.x;
    if (i >= BV * EDV * NCH) return;
    int be = i % (BV * EDV), c = i / (BV * EDV);
    int e = be % EDV, b = be / EDV;
    float Ae[NSV], p[NSV], h[NSV];
    #pragma unroll
    for (int n = 0; n < NSV; ++n) { Ae[n] = -__expf(Alog[e * NSV + n]); p[n] = 1.0f; h[n] = 0.0f; }
    for (int t = c * LCH; t < (c + 1) * LCH; ++t) {
        size_t bt = (size_t)b * TV + t;
        float d  = delta[bt * EDV + e];
        float xv = xin[bt * EDV + e];
        const float* BC = dbc + bt * FV + DTR;
        float dx = d * xv;
        #pragma unroll
        for (int n = 0; n < NSV; ++n) {
            float an = __expf(d * Ae[n]);
            p[n] *= an;
            h[n] = an * h[n] + dx * BC[n];
        }
    }
    float* Pp = P + (size_t)i * NSV;
    float* Sp = S + (size_t)i * NSV;
    #pragma unroll
    for (int n = 0; n < NSV; ++n) { Pp[n] = p[n]; Sp[n] = h[n]; }
}

__global__ __launch_bounds__(256)
void scan_phase2(const float* __restrict__ P, const float* __restrict__ S,
                 float* __restrict__ Hin) {
    int be = blockIdx.x * blockDim.x + threadIdx.x;
    if (be >= BV * EDV) return;
    float H[NSV];
    #pragma unroll
    for (int n = 0; n < NSV; ++n) H[n] = 0.0f;
    for (int c = 0; c < NCH; ++c) {
        size_t idx = ((size_t)c * (BV * EDV) + be) * NSV;
        #pragma unroll
        for (int n = 0; n < NSV; ++n) {
            Hin[idx + n] = H[n];
            H[n] = P[idx + n] * H[n] + S[idx + n];
        }
    }
}

__global__ __launch_bounds__(256)
void scan_phase3(const float* __restrict__ delta, const float* __restrict__ xin,
                 const float* __restrict__ dbc, const float* __restrict__ xz,
                 const float* __restrict__ Alog, const float* __restrict__ Dp,
                 const float* __restrict__ Hin, float* __restrict__ yg) {
    int i = blockIdx.x * blockDim.x + threadIdx.x;
    if (i >= BV * EDV * NCH) return;
    int be = i % (BV * EDV), c = i / (BV * EDV);
    int e = be % EDV, b = be / EDV;
    float Ae[NSV], h[NSV];
    #pragma unroll
    for (int n = 0; n < NSV; ++n) {
        Ae[n] = -__expf(Alog[e * NSV + n]);
        h[n]  = Hin[(size_t)i * NSV + n];
    }
    const float De = Dp[e];
    for (int t = c * LCH; t < (c + 1) * LCH; ++t) {
        size_t bt = (size_t)b * TV + t;
        float d  = delta[bt * EDV + e];
        float xv = xin[bt * EDV + e];
        const float* BC = dbc + bt * FV + DTR;
        float dx = d * xv, y = 0.0f;
        #pragma unroll
        for (int n = 0; n < NSV; ++n) {
            h[n] = __expf(d * Ae[n]) * h[n] + dx * BC[n];
            y += h[n] * BC[NSV + n];
        }
        y += De * xv;
        float z = xz[bt * (2 * EDV) + EDV + e];
        yg[bt * EDV + e] = y * (z * sigmoidf_(z));
    }
}

// LayerNorm over last dim (768), one block per row
__global__ __launch_bounds__(256)
void ln_kernel(const float* __restrict__ in, const float* __restrict__ w,
               const float* __restrict__ b, float* __restrict__ out) {
    __shared__ float red[256];
    const int row = blockIdx.x;
    const float* xr = in + (size_t)row * DV;
    float s = 0.0f;
    for (int d = threadIdx.x; d < DV; d += 256) s += xr[d];
    red[threadIdx.x] = s; __syncthreads();
    for (int o = 128; o > 0; o >>= 1) { if (threadIdx.x < o) red[threadIdx.x] += red[threadIdx.x + o]; __syncthreads(); }
    const float mu = red[0] / DV;
    __syncthreads();
    float v = 0.0f;
    for (int d = threadIdx.x; d < DV; d += 256) { float t = xr[d] - mu; v += t * t; }
    red[threadIdx.x] = v; __syncthreads();
    for (int o = 128; o > 0; o >>= 1) { if (threadIdx.x < o) red[threadIdx.x] += red[threadIdx.x + o]; __syncthreads(); }
    const float rs = rsqrtf(red[0] / DV + 1e-5f);
    for (int d = threadIdx.x; d < DV; d += 256)
        out[(size_t)row * DV + d] = (xr[d] - mu) * rs * w[d] + b[d];
}

// ==========================================================================
// Host orchestration
// ==========================================================================
extern "C" void kernel_launch(void* const* d_in, const int* in_sizes, int n_in,
                              void* d_out, int out_size, void* d_ws, size_t ws_size,
                              hipStream_t stream) {
    const int*   tokens = (const int*)  d_in[0];
    const float* embedw = (const float*)d_in[1];
    const float* in_w   = (const float*)d_in[2];
    const float* conv_w = (const float*)d_in[3];
    const float* conv_b = (const float*)d_in[4];
    const float* xp_w   = (const float*)d_in[5];
    const float* dt_w   = (const float*)d_in[6];
    const float* dt_b   = (const float*)d_in[7];
    const float* A_log  = (const float*)d_in[8];
    const float* D_par  = (const float*)d_in[9];
    const float* out_w  = (const float*)d_in[10];
    const float* ln_w   = (const float*)d_in[11];
    const float* ln_b   = (const float*)d_in[12];
    const float* head_w = (const float*)d_in[13];
    const float* head_b = (const float*)d_in[14];
    float* out = (float*)d_out;

    char*  base = (char*)d_ws;
    size_t off  = 0;
    auto carve = [&](size_t bytes) -> float* {
        float* p = (float*)(base + off);
        off = (off + bytes + 255) & ~(size_t)255;
        return p;
    };
    float* x     = carve((size_t)BT * DV      * 4);
    float* xz    = carve((size_t)BT * 2 * EDV * 4);
    float* xin   = carve((size_t)BT * EDV     * 4);
    float* dbc   = carve((size_t)BT * FV      * 4);
    float* delta = carve((size_t)BT * EDV     * 4);
    float* yg    = carve((size_t)BT * EDV     * 4);
    float* preln = carve((size_t)BT * DV      * 4);
    float* Pbuf  = carve((size_t)BV * EDV * NCH * NSV * 4);
    float* Sbuf  = carve((size_t)BV * EDV * NCH * NSV * 4);
    float* Hin   = carve((size_t)BV * EDV * NCH * NSV * 4);
    (void)ws_size; (void)n_in; (void)in_sizes; (void)out_size;

    embed_kernel<<<(BT * DV + 255) / 256, 256, 0, stream>>>(tokens, embedw, x, BT * DV);

    for (int l = 0; l < NLV; ++l) {
        mamba_gemm_wmma_bf16<<<dim3(BT / GBM, (2 * EDV) / GBN), 256, 0, stream>>>(
            x, in_w + (size_t)l * 2 * EDV * DV, nullptr, xz, BT, 2 * EDV, DV, 0);

        conv_silu_kernel<<<(BT * EDV + 255) / 256, 256, 0, stream>>>(
            xz, conv_w + (size_t)l * EDV * CK, conv_b + (size_t)l * EDV, xin);

        xproj_kernel<<<(BT * FV + 255) / 256, 256, 0, stream>>>(
            xin, xp_w + (size_t)l * FV * EDV, dbc);

        dtproj_kernel<<<(BT * EDV + 255) / 256, 256, 0, stream>>>(
            dbc, dt_w + (size_t)l * EDV * DTR, dt_b + (size_t)l * EDV, delta);

        scan_phase1<<<(BV * EDV * NCH + 255) / 256, 256, 0, stream>>>(
            delta, xin, dbc, A_log + (size_t)l * EDV * NSV, Pbuf, Sbuf);
        scan_phase2<<<(BV * EDV + 255) / 256, 256, 0, stream>>>(Pbuf, Sbuf, Hin);
        scan_phase3<<<(BV * EDV * NCH + 255) / 256, 256, 0, stream>>>(
            delta, xin, dbc, xz, A_log + (size_t)l * EDV * NSV,
            D_par + (size_t)l * EDV, Hin, yg);

        mamba_gemm_wmma_bf16<<<dim3(BT / GBM, DV / GBN), 256, 0, stream>>>(
            yg, out_w + (size_t)l * DV * EDV, nullptr, preln, BT, DV, EDV, 0);

        ln_kernel<<<BT, 256, 0, stream>>>(preln, ln_w + (size_t)l * DV, ln_b + (size_t)l * DV, x);
    }

    // logits = x @ head_w^T + head_b ; streaming output -> non-temporal stores
    mamba_gemm_wmma_bf16<<<dim3(BT / GBM, VV / GBN), 256, 0, stream>>>(
        x, head_w, head_b, out, BT, VV, DV, 1);
}
